// MultiHeadAttention_24953759990250
// MI455X (gfx1250) — matrix operand order repaired and checked by reading, compiled, not run on hardware
//
#include <hip/hip_runtime.h>

typedef __attribute__((ext_vector_type(16))) _Float16 v16h;
typedef __attribute__((ext_vector_type(8)))  _Float16 v8h;
typedef __attribute__((ext_vector_type(4)))  _Float16 v4h;
typedef __attribute__((ext_vector_type(8)))  float    v8f;
typedef __attribute__((ext_vector_type(4)))  float    v4f;

static constexpr int S  = 2048;   // sequence length
static constexpr int E  = 1024;   // embed dim
static constexpr int H  = 16;     // heads
static constexpr int DH = 64;     // head dim

// ---------------------------------------------------------------------------
// Fragment helpers per CDNA5 ISA 7.12.2 (wave32; 16-bit A 16x32, B 32x16,
// f32 C/D 16x16).  All f16 fragment loads are contiguous b128s.
// ---------------------------------------------------------------------------
__device__ __forceinline__ v16h join16(v8h lo, v8h hi) {
  v16h a;
#pragma unroll
  for (int i = 0; i < 8; ++i) { a[i] = lo[i]; a[8 + i] = hi[i]; }
  return a;
}

// A fragment from f16 row-major source (rows pre-offset to m0).
// lane<16: K = 16*(e/8)+e%8 ; lane>=16: +8.
__device__ __forceinline__ v16h load_a_f16(const _Float16* __restrict__ A,
                                           int lda, int k0) {
  const int lane = threadIdx.x & 31;
  const int g    = lane >> 4;
  const int row  = lane & 15;
  const _Float16* p = A + (long)row * lda + k0 + g * 8;
  return join16(*(const v8h*)p, *(const v8h*)(p + 16));
}

// B fragment from TRANSPOSED f16 source Bt[N][K]: b[e] = Bt[n0+col][k0+16*(e/8)+g*8+e%8].
__device__ __forceinline__ v16h load_bt_f16(const _Float16* __restrict__ Bt,
                                            int ldb, int k0, int n0) {
  const int lane = threadIdx.x & 31;
  const int g    = lane >> 4;
  const int col  = lane & 15;
  const _Float16* p = Bt + (long)(n0 + col) * ldb + k0 + g * 8;
  return join16(*(const v8h*)p, *(const v8h*)(p + 16));
}

// A fragment from f32 source with convert (used for the attn matrix only).
__device__ __forceinline__ v16h load_a_f32cvt(const float* __restrict__ A,
                                              int lda, int k0) {
  const int lane = threadIdx.x & 31;
  const int g    = lane >> 4;
  const int row  = lane & 15;
  const float* p = A + (long)row * lda + k0 + g * 8;
  const v4f f0 = *(const v4f*)p,        f1 = *(const v4f*)(p + 4);
  const v4f f2 = *(const v4f*)(p + 16), f3 = *(const v4f*)(p + 20);
  v16h a;
#pragma unroll
  for (int i = 0; i < 4; ++i) {
    a[i] = (_Float16)f0[i]; a[4 + i]  = (_Float16)f1[i];
    a[8 + i] = (_Float16)f2[i]; a[12 + i] = (_Float16)f3[i];
  }
  return a;
}

__device__ __forceinline__ v8f wmma16x16x32(v16h a, v16h b, v8f c) {
  return __builtin_amdgcn_wmma_f32_16x16x32_f16(false, a, false, b, (short)0, c,
                                                false, false);
}

// ---------------------------------------------------------------------------
// Elementwise f32 -> f16 convert.  n4 = element count / 4.
// ---------------------------------------------------------------------------
__global__ void convert_f16_kernel(const float* __restrict__ X,
                                   _Float16* __restrict__ Y, int n4) {
  const long i = (long)blockIdx.x * blockDim.x + threadIdx.x;
  if (i >= n4) return;
  const v4f v = *(const v4f*)(X + 4 * i);
  v4h h;
#pragma unroll
  for (int j = 0; j < 4; ++j) h[j] = (_Float16)v[j];
  *(v4h*)(Y + 4 * i) = h;
}

// ---------------------------------------------------------------------------
// Tiled transpose + convert: W[E][E] f32 -> Wt[E][E] f16, Wt[n][k] = W[k][n].
// grid = (E/32, E/32), block = (32, 8).
// ---------------------------------------------------------------------------
__global__ void transpose_f16_kernel(const float* __restrict__ W,
                                     _Float16* __restrict__ Wt) {
  __shared__ float tile[32][33];
  const int tx = threadIdx.x, ty = threadIdx.y;
  const int k0 = blockIdx.y * 32, n0 = blockIdx.x * 32;
#pragma unroll
  for (int i = 0; i < 4; ++i)
    tile[ty + 8 * i][tx] = W[(long)(k0 + ty + 8 * i) * E + n0 + tx];
  __syncthreads();
#pragma unroll
  for (int i = 0; i < 4; ++i)
    Wt[(long)(n0 + ty + 8 * i) * E + k0 + tx] = (_Float16)tile[tx][ty + 8 * i];
}

// ---------------------------------------------------------------------------
// f16 GEMM: C[M,N] = A[M,K] @ Bt[N,K]^T.  Each wave owns four 16x16 tiles
// along M sharing one B fragment; k-loop unrolled x2 -> 8 WMMAs per body.
// MODE 0: store f16 C row-major.   MODE 1: store f16 C^T (packed b128).
// MODE 2: store f32 C + f32 residual.
// grid = (N/64, M/64), block = 128.
// ---------------------------------------------------------------------------
template <int MODE>
__global__ void gemm_f16_kernel(const _Float16* __restrict__ A, int lda,
                                const _Float16* __restrict__ Bt, int ldb,
                                void* __restrict__ Cv, int ldc, int K,
                                const float* __restrict__ res, int ldr) {
  const int wave = threadIdx.x >> 5;
  const int n0   = blockIdx.x * 64 + wave * 16;
  const int m0   = blockIdx.y * 64;
  v8f acc[4] = {{}, {}, {}, {}};
#pragma unroll 2
  for (int k0 = 0; k0 < K; k0 += 32) {
    const v16h b = load_bt_f16(Bt, ldb, k0, n0);
#pragma unroll
    for (int i = 0; i < 4; ++i) {
      const v16h a = load_a_f16(A + (long)(m0 + 16 * i) * lda, lda, k0);
      acc[i] = wmma16x16x32(a, b, acc[i]);
    }
  }
  const int lane = threadIdx.x & 31;
  const int g = lane >> 4, col = lane & 15;
  if constexpr (MODE == 0) {
    _Float16* C = (_Float16*)Cv;
#pragma unroll
    for (int i = 0; i < 4; ++i)
#pragma unroll
      for (int r = 0; r < 8; ++r)
        C[(long)(m0 + 16 * i + r + 8 * g) * ldc + n0 + col] = (_Float16)acc[i][r];
  } else if constexpr (MODE == 1) {
    _Float16* Ct = (_Float16*)Cv;                  // Ct[N][M], ldc = M stride
#pragma unroll
    for (int i = 0; i < 4; ++i) {
      v8h v;
#pragma unroll
      for (int r = 0; r < 8; ++r) v[r] = (_Float16)acc[i][r];
      *(v8h*)(Ct + (long)(n0 + col) * ldc + m0 + 16 * i + 8 * g) = v;
    }
  } else {
    float* C = (float*)Cv;
#pragma unroll
    for (int i = 0; i < 4; ++i)
#pragma unroll
      for (int r = 0; r < 8; ++r) {
        const int mm = m0 + 16 * i + r + 8 * g;
        C[(long)mm * ldc + n0 + col] = acc[i][r] + res[(long)mm * ldr + n0 + col];
      }
  }
}

// ---------------------------------------------------------------------------
// Raw causal scores: attn[h][s][t] = (Q_h . K_h^T)/8 for tiles touching the
// lower triangle; strictly-upper tiles skipped (softmax pass writes zeros).
// Q, Km are f16 [S][E].  grid = (S/16, S/64, H), block = 128.
// ---------------------------------------------------------------------------
__global__ void scores_kernel(const _Float16* __restrict__ Q,
                              const _Float16* __restrict__ Km,
                              float* __restrict__ attn) {
  const int wave = threadIdx.x >> 5;
  const int h    = blockIdx.z;
  const int n0   = blockIdx.x * 16;                 // t
  const int m0   = (blockIdx.y * 4 + wave) * 16;    // s
  if (n0 > m0 + 15) return;                         // fully above diagonal
  const _Float16* Qh = Q + h * DH + (long)m0 * E;
  const _Float16* Kh = Km + h * DH;
  v8f acc = {};
#pragma unroll
  for (int k0 = 0; k0 < DH; k0 += 32) {
    const v16h a = load_a_f16(Qh, E, k0);
    const v16h b = load_bt_f16(Kh, E, k0, n0);      // Km rows are the "N" dim
    acc = wmma16x16x32(a, b, acc);
  }
  float* o = attn + (long)h * S * S;
  const int lane = threadIdx.x & 31;
  const int g = lane >> 4, col = lane & 15;
#pragma unroll
  for (int r = 0; r < 8; ++r)
    o[(long)(m0 + r + 8 * g) * S + n0 + col] = acc[r] * 0.125f;  // 1/sqrt(64)
}

// ---------------------------------------------------------------------------
// tril(softmax)/rowsum == exp(x - max_{t<=s}) / sum_{t<=s} exp(.)  (the full-
// row softmax denominator cancels).  One block per (h, s) row.
// ---------------------------------------------------------------------------
__global__ void softmax_tril_kernel(float* __restrict__ attn) {
  const int  tid = threadIdx.x;
  const long row = blockIdx.x;
  const int  s   = (int)(row & (S - 1));
  float* p = attn + row * (long)S;
  const int n = s + 1;
  __shared__ float sbuf[256];

  float m = -3.402823466e38f;
  for (int t = tid; t < n; t += 256) m = fmaxf(m, p[t]);
  sbuf[tid] = m; __syncthreads();
  for (int off = 128; off; off >>= 1) {
    if (tid < off) sbuf[tid] = fmaxf(sbuf[tid], sbuf[tid + off]);
    __syncthreads();
  }
  m = sbuf[0];
  __syncthreads();

  float sum = 0.f;
  for (int t = tid; t < n; t += 256) {
    const float e = __expf(p[t] - m);
    p[t] = e;
    sum += e;
  }
  sbuf[tid] = sum; __syncthreads();
  for (int off = 128; off; off >>= 1) {
    if (tid < off) sbuf[tid] += sbuf[tid + off];
    __syncthreads();
  }
  const float inv = 1.f / sbuf[0];
  for (int t4 = tid * 4; t4 < S; t4 += 1024) {      // b128 stores
    v4f v;
#pragma unroll
    for (int j = 0; j < 4; ++j) {
      const int t = t4 + j;
      v[j] = (t < n) ? p[t] * inv : 0.f;
    }
    *(v4f*)(p + t4) = v;
  }
}

// ---------------------------------------------------------------------------
// WS[s][h*64+d] = sum_{t<=s} attn[h][s][t] * V^T[h*64+d][t].  Each wave: four
// 16x16 tiles along s sharing one V fragment; k-loop rounded up to the 64-row
// block (attn is exactly zero beyond the diagonal).  Vt is f16 [E][S]; WS is
// f16 [S][E].  grid = (1, S/64, H), block = 128 (4 waves cover DH).
// ---------------------------------------------------------------------------
__global__ void av_kernel(const float* __restrict__ attn,
                          const _Float16* __restrict__ Vt,
                          _Float16* __restrict__ WS) {
  const int wave = threadIdx.x >> 5;
  const int h    = blockIdx.z;
  const int n0   = wave * 16;                       // d within head
  const int m0   = blockIdx.y * 64;                 // s block
  const float* Ah = attn + (long)h * S * S;
  const _Float16* Bt = Vt + (long)h * DH * S;       // rows = d, cols = t
  v8f acc[4] = {{}, {}, {}, {}};
  const int kend = m0 + 64;
#pragma unroll 2
  for (int k0 = 0; k0 < kend; k0 += 32) {
    const v16h b = load_bt_f16(Bt, S, k0, n0);
#pragma unroll
    for (int i = 0; i < 4; ++i) {
      const v16h a = load_a_f32cvt(Ah + (long)(m0 + 16 * i) * S, S, k0);
      acc[i] = wmma16x16x32(a, b, acc[i]);
    }
  }
  const int lane = threadIdx.x & 31;
  const int g = lane >> 4, col = lane & 15;
#pragma unroll
  for (int i = 0; i < 4; ++i)
#pragma unroll
    for (int r = 0; r < 8; ++r)
      WS[(long)(m0 + 16 * i + r + 8 * g) * E + h * DH + n0 + col] =
          (_Float16)acc[i][r];
}

// ---------------------------------------------------------------------------
// LayerNorm over last dim. One block per row. grid = S, block = 256.
// ---------------------------------------------------------------------------
__global__ void layernorm_kernel(const float* __restrict__ X,
                                 const float* __restrict__ gamma,
                                 const float* __restrict__ beta,
                                 float* __restrict__ out) {
  const int tid = threadIdx.x;
  const int s   = blockIdx.x;
  const float* x = X + (long)s * E;
  __shared__ float s1[256];
  __shared__ float s2[256];
  float a = 0.f, b = 0.f;
  for (int t = tid; t < E; t += 256) { const float v = x[t]; a += v; b += v * v; }
  s1[tid] = a; s2[tid] = b; __syncthreads();
  for (int off = 128; off; off >>= 1) {
    if (tid < off) { s1[tid] += s1[tid + off]; s2[tid] += s2[tid + off]; }
    __syncthreads();
  }
  const float mean = s1[0] * (1.f / E);
  const float var  = s2[0] * (1.f / E) - mean * mean;
  const float inv  = rsqrtf(var + 1e-5f);
  float* o = out + (long)s * E;
  for (int t = tid; t < E; t += 256)
    o[t] = (x[t] - mean) * inv * gamma[t] + beta[t];
}

// ---------------------------------------------------------------------------
extern "C" void kernel_launch(void* const* d_in, const int* in_sizes, int n_in,
                              void* d_out, int out_size, void* d_ws, size_t ws_size,
                              hipStream_t stream) {
  const float* query = (const float*)d_in[0];
  const float* key   = (const float*)d_in[1];
  const float* value = (const float*)d_in[2];
  const float* Wq    = (const float*)d_in[3];
  const float* Wk    = (const float*)d_in[4];
  const float* Wv    = (const float*)d_in[5];
  const float* Wo    = (const float*)d_in[6];
  const float* lnw   = (const float*)d_in[7];
  const float* lnb   = (const float*)d_in[8];

  float* out  = (float*)d_out;                 // [S, E]
  float* attn = out + (size_t)S * E;           // [H, S, S]

  // Workspace (44 MB).
  _Float16* qh  = (_Float16*)d_ws;             // f16 activations [S,E]
  _Float16* kh  = qh + (size_t)S * E;
  _Float16* vh  = kh + (size_t)S * E;
  _Float16* Wqt = vh + (size_t)S * E;          // f16 transposed weights [E,E]
  _Float16* Wkt = Wqt + (size_t)E * E;
  _Float16* Wvt = Wkt + (size_t)E * E;
  _Float16* Wot = Wvt + (size_t)E * E;
  _Float16* Qp  = Wot + (size_t)E * E;         // f16 [S,E]
  _Float16* Kp  = Qp + (size_t)S * E;          // f16 [S,E]
  _Float16* Vt  = Kp + (size_t)S * E;          // f16 [E,S] (transposed V)
  _Float16* WSp = Vt + (size_t)S * E;          // f16 [S,E]
  float*    Tmp = (float*)(WSp + (size_t)S * E);  // f32 [S,E] pre-LN

  const dim3 blk(128);
  const dim3 ggrid(E / 64, S / 64);
  const int  n4 = S * E / 4;

  // 0) Materialize f16 operands (contiguous b128 fragment loads downstream).
  convert_f16_kernel<<<dim3(n4 / 256), dim3(256), 0, stream>>>(query, qh, n4);
  convert_f16_kernel<<<dim3(n4 / 256), dim3(256), 0, stream>>>(key,   kh, n4);
  convert_f16_kernel<<<dim3(n4 / 256), dim3(256), 0, stream>>>(value, vh, n4);
  const dim3 tgrid(E / 32, E / 32), tblk(32, 8);
  transpose_f16_kernel<<<tgrid, tblk, 0, stream>>>(Wq, Wqt);
  transpose_f16_kernel<<<tgrid, tblk, 0, stream>>>(Wk, Wkt);
  transpose_f16_kernel<<<tgrid, tblk, 0, stream>>>(Wv, Wvt);
  transpose_f16_kernel<<<tgrid, tblk, 0, stream>>>(Wo, Wot);

  // 1) Projections (V stored transposed for contiguous B fragments later).
  gemm_f16_kernel<0><<<ggrid, blk, 0, stream>>>(qh, E, Wqt, E, Qp, E, E, nullptr, 0);
  gemm_f16_kernel<0><<<ggrid, blk, 0, stream>>>(kh, E, Wkt, E, Kp, E, E, nullptr, 0);
  gemm_f16_kernel<1><<<ggrid, blk, 0, stream>>>(vh, E, Wvt, E, Vt, S, E, nullptr, 0);

  // 2) Causal raw scores.
  scores_kernel<<<dim3(S / 16, S / 64, H), blk, 0, stream>>>(Qp, Kp, attn);

  // 3) Softmax + tril + renormalize (writes the full attn tensor).
  softmax_tril_kernel<<<dim3(H * S), dim3(256), 0, stream>>>(attn);

  // 4) attn @ V -> WS (f16).
  av_kernel<<<dim3(1, S / 64, H), blk, 0, stream>>>(attn, Vt, WSp);

  // 5) Output projection + residual -> Tmp (f32).
  gemm_f16_kernel<2><<<ggrid, blk, 0, stream>>>(WSp, E, Wot, E, Tmp, E, E, query, E);

  // 6) LayerNorm -> out.
  layernorm_kernel<<<dim3(S), dim3(256), 0, stream>>>(Tmp, lnw, lnb, out);
}
